// MultiHeadAttention_29059748725165
// MI455X (gfx1250) — matrix operand order repaired and checked by reading, not compiled, not run on hardware
//
#include <hip/hip_runtime.h>
#include <cstdint>

// ---------------- types for WMMA fragments ----------------
typedef __bf16 bf16_t;
typedef __attribute__((ext_vector_type(16))) __bf16 v16bf;
typedef __attribute__((ext_vector_type(8)))  __bf16 v8bf;
typedef __attribute__((ext_vector_type(8)))  float  v8f;

union ABFrag { v16bf v; v8bf h[2]; };

// ---------------- helpers ----------------
__global__ void f32_to_bf16_kernel(const float* __restrict__ in, bf16_t* __restrict__ out, int n)
{
    int i = blockIdx.x * blockDim.x + threadIdx.x;
    if (i < n) out[i] = (bf16_t)in[i];
}

// W is [K x N] row-major fp32 -> WT is [N x K] row-major bf16
__global__ void transpose_f32_to_bf16_kernel(const float* __restrict__ W, bf16_t* __restrict__ WT,
                                             int K, int N)
{
    __shared__ float tile[32][33];
    const int n0 = blockIdx.x * 32;
    const int k0 = blockIdx.y * 32;
    const int tx = threadIdx.x;   // 0..31
    const int ty = threadIdx.y;   // 0..7
#pragma unroll
    for (int j = 0; j < 32; j += 8)
        tile[ty + j][tx] = W[(size_t)(k0 + ty + j) * N + (n0 + tx)];
    __syncthreads();
#pragma unroll
    for (int j = 0; j < 32; j += 8)
        WT[(size_t)(n0 + ty + j) * K + (k0 + tx)] = (bf16_t)tile[tx][ty + j];
}

// ---------------- WMMA GEMM ----------------
// C[M,N] = A[M,K] @ (BT[N,K])^T (+ bias[n]) with epilogue variants.
// A, BT bf16 row-major with leading stride K (true for every use below).
// MODE 0: store bf16 row-major
// MODE 1: store bf16 transposed per batch: C[b][n][s], b = row/Sbatch
// MODE 2: store f32 row-major (+bias)
// MODE 3: scores: f32 row-major, val = acc*scale + (n>m ? -1e9 : 0)
template<int MODE>
__device__ inline void store_tile(v8f acc, void* Cv, const float* bias,
                                  int tileM, int n, int half, int N, int Sbatch, float scale)
{
    if constexpr (MODE == 0) {
        bf16_t* C = (bf16_t*)Cv;
        const float bv = bias ? bias[n] : 0.f;
#pragma unroll
        for (int e = 0; e < 8; ++e) {
            const int m = tileM + half * 8 + e;
            C[(size_t)m * N + n] = (bf16_t)(acc[e] + bv);
        }
    } else if constexpr (MODE == 1) {
        bf16_t* C = (bf16_t*)Cv;
        const int b  = tileM / Sbatch;
        const int mm = tileM % Sbatch;
        const float bv = bias ? bias[n] : 0.f;
        v8bf outv;
#pragma unroll
        for (int e = 0; e < 8; ++e) outv[e] = (bf16_t)(acc[e] + bv);
        *(v8bf*)(C + (size_t)b * N * Sbatch + (size_t)n * Sbatch + mm + half * 8) = outv;
    } else if constexpr (MODE == 2) {
        float* C = (float*)Cv;
        const float bv = bias ? bias[n] : 0.f;
#pragma unroll
        for (int e = 0; e < 8; ++e) {
            const int m = tileM + half * 8 + e;
            C[(size_t)m * N + n] = acc[e] + bv;
        }
    } else { // MODE == 3
        float* C = (float*)Cv;
#pragma unroll
        for (int e = 0; e < 8; ++e) {
            const int m = tileM + half * 8 + e;           // query index s
            C[(size_t)m * N + n] = acc[e] * scale + ((n > m) ? -1e9f : 0.f);
        }
    }
}

template<int MODE>
__global__ void __launch_bounds__(32)
gemm16_wmma(const bf16_t* __restrict__ A, const bf16_t* __restrict__ BT,
            const float* __restrict__ bias, void* __restrict__ Cv,
            int M, int N, int K, int Sbatch, float scale)
{
    const int lane  = threadIdx.x;     // 0..31 (wave32)
    const int half  = lane >> 4;       // 0 | 1
    const int l16   = lane & 15;
    const int tileM  = blockIdx.y * 16;
    const int tileN0 = blockIdx.x * 64;

    // Per-lane base pointers (A frag: rows, two 8-elt runs; B frag: two 8-elt runs)
    const bf16_t* __restrict__ Arow = A + (size_t)(tileM + l16) * K + half * 8;
    const bf16_t* __restrict__ B0   = BT + (size_t)(tileN0 + l16) * K + half * 8;
    const bf16_t* __restrict__ B1   = B0 + (size_t)16 * K;
    const bf16_t* __restrict__ B2   = B0 + (size_t)32 * K;
    const bf16_t* __restrict__ B3   = B0 + (size_t)48 * K;

    v8f acc0 = {}, acc1 = {}, acc2 = {}, acc3 = {};

    // Ping-pong fragment buffers for software pipelining
    ABFrag aA, aB, b0A, b1A, b2A, b3A, b0B, b1B, b2B, b3B;

#define LOAD_SET(SUF, K0)                                                          \
    {                                                                              \
        a##SUF.h[0]  = *(const v8bf*)(Arow + (K0));                                \
        a##SUF.h[1]  = *(const v8bf*)(Arow + (K0) + 16);                           \
        b0##SUF.h[0] = *(const v8bf*)(B0 + (K0));                                  \
        b0##SUF.h[1] = *(const v8bf*)(B0 + (K0) + 16);                             \
        b1##SUF.h[0] = *(const v8bf*)(B1 + (K0));                                  \
        b1##SUF.h[1] = *(const v8bf*)(B1 + (K0) + 16);                             \
        b2##SUF.h[0] = *(const v8bf*)(B2 + (K0));                                  \
        b2##SUF.h[1] = *(const v8bf*)(B2 + (K0) + 16);                             \
        b3##SUF.h[0] = *(const v8bf*)(B3 + (K0));                                  \
        b3##SUF.h[1] = *(const v8bf*)(B3 + (K0) + 16);                             \
    }

#define MMA_SET(SUF)                                                               \
    {                                                                              \
        acc0 = __builtin_amdgcn_wmma_f32_16x16x32_bf16(                            \
                   false, a##SUF.v, false, b0##SUF.v, (short)0, acc0, false, false);\
        acc1 = __builtin_amdgcn_wmma_f32_16x16x32_bf16(                            \
                   false, a##SUF.v, false, b1##SUF.v, (short)0, acc1, false, false);\
        acc2 = __builtin_amdgcn_wmma_f32_16x16x32_bf16(                            \
                   false, a##SUF.v, false, b2##SUF.v, (short)0, acc2, false, false);\
        acc3 = __builtin_amdgcn_wmma_f32_16x16x32_bf16(                            \
                   false, a##SUF.v, false, b3##SUF.v, (short)0, acc3, false, false);\
    }

    // K is a multiple of 64 for every GEMM in this model (1024 or 2048).
    LOAD_SET(A, 0)
    int k0 = 0;
    for (; k0 < K - 64; k0 += 64) {
        LOAD_SET(B, k0 + 32)      // issue next-half loads before consuming A-buffer
        MMA_SET(A)
        LOAD_SET(A, k0 + 64)      // issue next-step loads before consuming B-buffer
        MMA_SET(B)
    }
    LOAD_SET(B, k0 + 32)
    MMA_SET(A)
    MMA_SET(B)

#undef LOAD_SET
#undef MMA_SET

    store_tile<MODE>(acc0, Cv, bias, tileM, tileN0 +  0 + l16, half, N, Sbatch, scale);
    store_tile<MODE>(acc1, Cv, bias, tileM, tileN0 + 16 + l16, half, N, Sbatch, scale);
    store_tile<MODE>(acc2, Cv, bias, tileM, tileN0 + 32 + l16, half, N, Sbatch, scale);
    store_tile<MODE>(acc3, Cv, bias, tileM, tileN0 + 48 + l16, half, N, Sbatch, scale);
}

// ---------------- row softmax: fp32 scores -> bf16 probabilities ----------------
__global__ void __launch_bounds__(256)
softmax_rows_kernel(const float* __restrict__ S, bf16_t* __restrict__ P, int n)
{
    const int row = blockIdx.x;
    const int tid = threadIdx.x;
    const float* x = S + (size_t)row * n;
    bf16_t*      p = P + (size_t)row * n;

    __shared__ float buf[256];

    float m = -3.4e38f;
    for (int i = tid; i < n; i += 256) m = fmaxf(m, x[i]);
    buf[tid] = m; __syncthreads();
    for (int off = 128; off > 0; off >>= 1) {
        if (tid < off) buf[tid] = fmaxf(buf[tid], buf[tid + off]);
        __syncthreads();
    }
    m = buf[0]; __syncthreads();

    float s = 0.f;
    for (int i = tid; i < n; i += 256) s += expf(x[i] - m);
    buf[tid] = s; __syncthreads();
    for (int off = 128; off > 0; off >>= 1) {
        if (tid < off) buf[tid] += buf[tid + off];
        __syncthreads();
    }
    s = buf[0];

    const float inv = 1.f / s;
    for (int i = tid; i < n; i += 256) p[i] = (bf16_t)(expf(x[i] - m) * inv);
}

// ---------------- host side ----------------
extern "C" void kernel_launch(void* const* d_in, const int* in_sizes, int n_in,
                              void* d_out, int out_size, void* d_ws, size_t ws_size,
                              hipStream_t stream)
{
    (void)in_sizes; (void)n_in; (void)out_size; (void)ws_size;

    constexpr int Bb = 4, S = 2048, D = 1024;
    constexpr int Mtot = Bb * S;                       // 8192
    const float scale = 0.35355339059327373f;          // 64^-0.25, applied once

    const float* q  = (const float*)d_in[0];
    // d_in[1] = mask (causal, regenerated analytically)
    const float* Wq = (const float*)d_in[2];
    const float* bq = (const float*)d_in[3];
    const float* Wk = (const float*)d_in[4];
    const float* Wv = (const float*)d_in[5];
    const float* bv = (const float*)d_in[6];
    const float* Wo = (const float*)d_in[7];
    const float* bo = (const float*)d_in[8];
    float* out = (float*)d_out;

    char* ws = (char*)d_ws;
    size_t off = 0;
    auto alloc = [&](size_t bytes) -> void* {
        void* p = ws + off;
        off += (bytes + 255) & ~size_t(255);
        return p;
    };

    bf16_t* qb  = (bf16_t*)alloc((size_t)Mtot * D * 2);   // q in bf16
    bf16_t* WqT = (bf16_t*)alloc((size_t)D * D * 2);
    bf16_t* WkT = (bf16_t*)alloc((size_t)D * D * 2);
    bf16_t* WvT = (bf16_t*)alloc((size_t)D * D * 2);
    bf16_t* WoT = (bf16_t*)alloc((size_t)D * D * 2);
    bf16_t* qp  = (bf16_t*)alloc((size_t)Mtot * D * 2);   // projected q
    bf16_t* kb  = (bf16_t*)alloc((size_t)Mtot * D * 2);   // k = qp @ Wk
    bf16_t* vT  = (bf16_t*)alloc((size_t)Mtot * D * 2);   // v transposed [B][D][S]
    bf16_t* wv  = (bf16_t*)alloc((size_t)Mtot * D * 2);   // attn @ v
    float*  sc  = (float*) alloc((size_t)S * S * 4);      // per-batch scores (L2-resident)
    bf16_t* P   = (bf16_t*)alloc((size_t)S * S * 2);      // per-batch probabilities

    // --- prepare bf16 operands ---
    f32_to_bf16_kernel<<<(Mtot * D) / 256, 256, 0, stream>>>(q, qb, Mtot * D);
    {
        dim3 tb(32, 8), tg(D / 32, D / 32);
        transpose_f32_to_bf16_kernel<<<tg, tb, 0, stream>>>(Wq, WqT, D, D);
        transpose_f32_to_bf16_kernel<<<tg, tb, 0, stream>>>(Wk, WkT, D, D);
        transpose_f32_to_bf16_kernel<<<tg, tb, 0, stream>>>(Wv, WvT, D, D);
        transpose_f32_to_bf16_kernel<<<tg, tb, 0, stream>>>(Wo, WoT, D, D);
    }

    // --- projections (all batches fused: rows never straddle batches) ---
    dim3 gProj(D / 64, Mtot / 16);
    gemm16_wmma<0><<<gProj, 32, 0, stream>>>(qb, WqT, bq,      qp, Mtot, D, D, S, 1.f);
    gemm16_wmma<0><<<gProj, 32, 0, stream>>>(qp, WkT, nullptr, kb, Mtot, D, D, S, 1.f);
    gemm16_wmma<1><<<gProj, 32, 0, stream>>>(qp, WvT, bv,      vT, Mtot, D, D, S, 1.f);

    // --- attention, one batch at a time (scores buffer reused, stays in L2) ---
    for (int b = 0; b < Bb; ++b) {
        const bf16_t* qp_b = qp + (size_t)b * S * D;
        const bf16_t* k_b  = kb + (size_t)b * S * D;
        const bf16_t* vT_b = vT + (size_t)b * D * S;
        bf16_t*       wv_b = wv + (size_t)b * S * D;

        dim3 gS(S / 64, S / 16);
        // scores: qp_b @ k_b^T * scale + causal(-1e9); k_b row-major IS B^T
        gemm16_wmma<3><<<gS, 32, 0, stream>>>(qp_b, k_b, nullptr, sc, S, S, D, S, scale);

        softmax_rows_kernel<<<S, 256, 0, stream>>>(sc, P, S);

        dim3 gPV(D / 64, S / 16);
        // wv = P @ v ; vT_b row-major [D][S] IS (v)^T
        gemm16_wmma<0><<<gPV, 32, 0, stream>>>(P, vT_b, nullptr, wv_b, S, D, S, S, 1.f);
    }

    // --- output projection ---
    gemm16_wmma<2><<<gProj, 32, 0, stream>>>(wv, WoT, bo, out, Mtot, D, D, S, 1.f);
}
